// SequenceModel_38585986187792
// MI455X (gfx1250) — matrix operand order repaired and checked by reading, not compiled, not run on hardware
//
#include <hip/hip_runtime.h>
#include <math.h>

// ---- model dims ----
#define BB   128
#define SS   256
#define DD   128
#define DIE  256            // mLSTM inner dim
#define NHH  4
#define DHMM 64             // mLSTM head dim
#define DHSS 32             // sLSTM head dim
#define KCONV 4
#define ROWS (BB*SS)        // 32768

typedef __attribute__((ext_vector_type(16))) _Float16 v16h;
typedef __attribute__((ext_vector_type(8)))  _Float16 v8h;
typedef __attribute__((ext_vector_type(8)))  float    v8f;
typedef unsigned int u32x4 __attribute__((ext_vector_type(4)));
typedef int          i32x4 __attribute__((ext_vector_type(4)));
typedef int          i32x8 __attribute__((ext_vector_type(8)));

__device__ __forceinline__ float dsilu(float x)  { return x / (1.0f + expf(-x)); }
__device__ __forceinline__ float dsigm(float x)  { return 1.0f / (1.0f + expf(-x)); }
__device__ __forceinline__ float dlogsig(float x){ return fminf(x, 0.0f) - log1pf(expf(-fabsf(x))); }
__device__ __forceinline__ float dgelu(float x)  { return 0.5f * x * (1.0f + erff(x * 0.70710678118654752f)); }

// ---------------------------------------------------------------------------
// Tensor Data Mover: 2D f16 tile (tile1 rows x tile0 cols) global -> LDS.
// D# per CDNA5 ISA ch.8: group0 {count=1, lds_addr, global_addr, type=2},
// group1 {data_size=1(2B), tensor_dim0/1, tile_dim0/1, dim0_stride}.
__device__ __forceinline__ void tdm_load_2d(unsigned ldsAddr, const void* gptr,
                                            unsigned dim0, unsigned dim1,
                                            unsigned stride0,
                                            unsigned tile0, unsigned tile1) {
  unsigned long long ga = (unsigned long long)(size_t)gptr;
  u32x4 g0;
  g0[0] = 1u;                                            // count=1, no gather
  g0[1] = ldsAddr;                                       // lds_addr (bytes)
  g0[2] = (unsigned)ga;                                  // global_addr[31:0]
  g0[3] = ((unsigned)(ga >> 32) & 0x01FFFFFFu) | (2u << 30); // addr[56:32], type=2
  i32x8 g1;
  g1[0] = 0x10000;                                       // data_size=1 (2 bytes)
  g1[1] = (int)(dim0 << 16);                             // tensor_dim0 -> bits[79:48]
  g1[2] = (int)((dim0 >> 16) | (dim1 << 16));            // tensor_dim1 -> bits[111:80]
  g1[3] = (int)((dim1 >> 16) | (tile0 << 16));           // tile_dim0 -> bits[127:112]
  g1[4] = (int)(tile1 & 0xFFFFu);                        // tile_dim1; tile_dim2=0
  g1[5] = (int)stride0;                                  // tensor_dim0_stride[31:0]
  g1[6] = 0;
  g1[7] = 0;
  i32x4 z4 = {0, 0, 0, 0};
#if __clang_major__ >= 23
  i32x8 z8 = {0, 0, 0, 0, 0, 0, 0, 0};
  __builtin_amdgcn_tensor_load_to_lds(g0, g1, z4, z4, z8, 0);
#else
  __builtin_amdgcn_tensor_load_to_lds(g0, g1, z4, z4, 0);
#endif
}

// ---------------------------------------------------------------------------
__global__ void k_embed(const int* __restrict__ x, const float* __restrict__ emb,
                        float* __restrict__ h) {
  int i = blockIdx.x * blockDim.x + threadIdx.x;
  if (i >= ROWS * DD) return;
  int row = i / DD, c = i - row * DD;
  h[i] = emb[x[row] * DD + c];
}

// ---------------------------------------------------------------------------
__global__ void k_ln(const float* __restrict__ x, const float* __restrict__ w,
                     float* __restrict__ y) {
  __shared__ float s1[DD], s2[DD];
  int row = blockIdx.x, t = threadIdx.x;
  float v = x[(size_t)row * DD + t];
  s1[t] = v; s2[t] = v * v;
  __syncthreads();
  for (int off = DD / 2; off > 0; off >>= 1) {
    if (t < off) { s1[t] += s1[t + off]; s2[t] += s2[t + off]; }
    __syncthreads();
  }
  float mu = s1[0] * (1.0f / DD);
  float var = s2[0] * (1.0f / DD) - mu * mu;
  y[(size_t)row * DD + t] = (v - mu) * rsqrtf(var + 1e-5f) * w[t];
}

// ---------------------------------------------------------------------------
// WMMA GEMM: C[M,N] (+)= A[M,K] @ W[N,K]^T.  One wave per 16x16 tile.
// Fragment chunks are contiguous 8/16-float runs -> float4 (b128) loads.
template <int ADDC>
__global__ void __launch_bounds__(256)
k_gemm(const float* __restrict__ A, const float* __restrict__ W,
       float* __restrict__ C, int M, int N, int Kd) {
  int wave = (blockIdx.x * blockDim.x + threadIdx.x) >> 5;
  int lane = threadIdx.x & 31;
  int tn = N >> 4;
  int tiles = (M >> 4) * tn;
  if (wave >= tiles) return;
  int tm = wave / tn;
  int tc = wave - tm * tn;
  int row = lane & 15, hi = lane >> 4;
  const float* aRow = A + (size_t)(tm * 16 + row) * Kd + hi * 8;
  const float* wRow = W + (size_t)(tc * 16 + row) * Kd + hi * 8;
  v8f acc = {0.f,0.f,0.f,0.f,0.f,0.f,0.f,0.f};
  for (int k0 = 0; k0 < Kd; k0 += 32) {
    float4 aA = *(const float4*)(aRow + k0);
    float4 aB = *(const float4*)(aRow + k0 + 4);
    float4 aC = *(const float4*)(aRow + k0 + 16);
    float4 aD = *(const float4*)(aRow + k0 + 20);
    float4 bA = *(const float4*)(wRow + k0);
    float4 bB = *(const float4*)(wRow + k0 + 4);
    float4 bC = *(const float4*)(wRow + k0 + 16);
    float4 bD = *(const float4*)(wRow + k0 + 20);
    v16h a, b;
    a[0]=(_Float16)aA.x; a[1]=(_Float16)aA.y; a[2]=(_Float16)aA.z; a[3]=(_Float16)aA.w;
    a[4]=(_Float16)aB.x; a[5]=(_Float16)aB.y; a[6]=(_Float16)aB.z; a[7]=(_Float16)aB.w;
    a[8]=(_Float16)aC.x; a[9]=(_Float16)aC.y; a[10]=(_Float16)aC.z; a[11]=(_Float16)aC.w;
    a[12]=(_Float16)aD.x; a[13]=(_Float16)aD.y; a[14]=(_Float16)aD.z; a[15]=(_Float16)aD.w;
    b[0]=(_Float16)bA.x; b[1]=(_Float16)bA.y; b[2]=(_Float16)bA.z; b[3]=(_Float16)bA.w;
    b[4]=(_Float16)bB.x; b[5]=(_Float16)bB.y; b[6]=(_Float16)bB.z; b[7]=(_Float16)bB.w;
    b[8]=(_Float16)bC.x; b[9]=(_Float16)bC.y; b[10]=(_Float16)bC.z; b[11]=(_Float16)bC.w;
    b[12]=(_Float16)bD.x; b[13]=(_Float16)bD.y; b[14]=(_Float16)bD.z; b[15]=(_Float16)bD.w;
    acc = __builtin_amdgcn_wmma_f32_16x16x32_f16(false, a, false, b,
                                                 (short)0, acc, false, false);
  }
  float* cp = C + (size_t)(tm * 16 + hi * 8) * N + tc * 16 + row;
#pragma unroll
  for (int v = 0; v < 8; ++v) {
    if (ADDC) cp[(size_t)v * N] += acc[v];
    else      cp[(size_t)v * N]  = acc[v];
  }
}

// ---------------------------------------------------------------------------
__global__ void k_conv_silu(const float* __restrict__ in, int ldin,
                            const float* __restrict__ w, const float* __restrict__ bias,
                            float* __restrict__ out, int C) {
  int i = blockIdx.x * blockDim.x + threadIdx.x;
  if (i >= ROWS * C) return;
  int row = i / C, c = i - row * C;
  int b = row / SS, s = row - b * SS;
  float acc = bias[c];
#pragma unroll
  for (int j = 0; j < KCONV; ++j) {
    int sp = s - (KCONV - 1) + j;
    if (sp >= 0) acc += in[(size_t)(b * SS + sp) * ldin + c] * w[c * KCONV + j];
  }
  out[(size_t)row * C + c] = dsilu(acc);
}

// ---------------------------------------------------------------------------
__global__ void k_blockdiag(const float* __restrict__ in, int ldin,
                            const float* __restrict__ w, int bs,
                            float* __restrict__ out, int C) {
  int i = blockIdx.x * blockDim.x + threadIdx.x;
  if (i >= ROWS * C) return;
  int row = i / C, c = i - row * C;
  int blk = c / bs, e = c - blk * bs;
  const float* wr = w + ((size_t)blk * bs + e) * bs;
  const float* xr = in + (size_t)row * ldin + blk * bs;
  float acc = 0.f;
  for (int d = 0; d < bs; ++d) acc += xr[d] * wr[d];
  out[(size_t)row * C + c] = acc;
}

// ---------------------------------------------------------------------------
__global__ void k_mgate(const float* __restrict__ q, const float* __restrict__ k,
                        const float* __restrict__ v,
                        const float* __restrict__ igw, const float* __restrict__ igb,
                        const float* __restrict__ fgw, const float* __restrict__ fgb,
                        float* __restrict__ ip, float* __restrict__ fp) {
  int i = blockIdx.x * blockDim.x + threadIdx.x;
  if (i >= ROWS * NHH * 2) return;
  int g = i & 1, t = i >> 1;
  int h = t & (NHH - 1), row = t >> 2;
  const float* w = (g ? fgw : igw) + (size_t)h * 3 * DIE;
  float acc = g ? fgb[h] : igb[h];
  const float* qr = q + (size_t)row * DIE;
  const float* kr = k + (size_t)row * DIE;
  const float* vr = v + (size_t)row * DIE;
  for (int c = 0; c < DIE; ++c)
    acc += qr[c] * w[c] + kr[c] * w[DIE + c] + vr[c] * w[2 * DIE + c];
  int b = row / SS, s = row - b * SS;
  size_t o = ((size_t)(b * NHH + h)) * SS + s;
  (g ? fp : ip)[o] = acc;
}

// ---------------------------------------------------------------------------
// f16 materialization for TDM: q16/k16 same layout, v transposed to (b,h,dh,s)
__global__ void k_tof16(const float* __restrict__ q, const float* __restrict__ k,
                        const float* __restrict__ v, _Float16* __restrict__ q16,
                        _Float16* __restrict__ k16, _Float16* __restrict__ vt) {
  int i = blockIdx.x * blockDim.x + threadIdx.x;
  if (i >= ROWS * DIE) return;
  q16[i] = (_Float16)q[i];
  k16[i] = (_Float16)k[i];
  int row = i / DIE, c = i - row * DIE;
  int b = row / SS, s = row - b * SS;
  int h = c >> 6, d = c & 63;
  vt[(((size_t)(b * NHH + h)) * DHMM + d) * SS + s] = (_Float16)v[i];
}

// ---------------------------------------------------------------------------
__global__ void k_mscan(const float* __restrict__ ip, const float* __restrict__ fp,
                        float* __restrict__ u, float* __restrict__ pm,
                        float* __restrict__ em) {
  int i = blockIdx.x * blockDim.x + threadIdx.x;
  if (i >= BB * NHH) return;
  size_t base = (size_t)i * SS;
  float lf = 0.f, p = -1e30f;
  for (int s = 0; s < SS; ++s) {
    lf += dlogsig(fp[base + s]);
    float uv = ip[base + s] - lf;
    p = fmaxf(p, uv);
    u[base + s] = uv; pm[base + s] = p;
    em[base + s] = expf(-(lf + p));
  }
}

// ---------------------------------------------------------------------------
// Causal mLSTM attention, one workgroup per (b,head). Q/K/V tiles arrive via
// Tensor Data Mover; QK^T and PV through WMMA; decay applied between.
__global__ void __launch_bounds__(256)
k_attn(const _Float16* __restrict__ q16, const _Float16* __restrict__ k16,
       const _Float16* __restrict__ vt, const float* __restrict__ u,
       const float* __restrict__ pm, const float* __restrict__ em,
       float* __restrict__ hm) {
  extern __shared__ char smem[];
  _Float16* qh = (_Float16*)smem;            // 256 x 64 (s-major)
  _Float16* kh = qh + SS * DHMM;             // 256 x 64 (s-major)
  _Float16* vh = kh + SS * DHMM;             // 64 x 256 (dh-major, transposed)
  float* uls = (float*)(vh + SS * DHMM);     // 256
  float* pls = uls + SS;
  float* els = pls + SS;
  _Float16* stg = (_Float16*)(els + SS);     // 8 waves * 16*32

  int bh = blockIdx.x, b = bh / NHH, h = bh - b * NHH;
  int tid = threadIdx.x;
  if (tid < 32) {                            // wave 0 issues the three DMAs
    size_t qkOff = (size_t)b * SS * DIE + h * DHMM;
    size_t vOff  = ((size_t)(b * NHH + h)) * DHMM * SS;
    tdm_load_2d((unsigned)(size_t)qh, q16 + qkOff, DHMM, SS, DIE, DHMM, SS);
    tdm_load_2d((unsigned)(size_t)kh, k16 + qkOff, DHMM, SS, DIE, DHMM, SS);
    tdm_load_2d((unsigned)(size_t)vh, vt + vOff,  SS, DHMM, SS, SS, DHMM);
    __builtin_amdgcn_s_wait_tensorcnt(0);
  }
  size_t gb = (size_t)bh * SS;
  for (int s = tid; s < SS; s += 256) {
    uls[s] = u[gb + s]; pls[s] = pm[gb + s]; els[s] = em[gb + s];
  }
  __syncthreads();

  int wid = tid >> 5, lane = tid & 31;
  int row = lane & 15, hi = lane >> 4;
  _Float16* mystg = stg + wid * 16 * 32;

  for (int qt = wid; qt < 16; qt += 8) {
    int q0 = qt * 16;
    v16h a0, a1;                             // q fragments, dh 0..31 / 32..63
    {
      const _Float16* qp = qh + (q0 + row) * DHMM + 8 * hi;
      v8h l0 = *(const v8h*)(qp);
      v8h h0 = *(const v8h*)(qp + 16);
      v8h l1 = *(const v8h*)(qp + 32);
      v8h h1 = *(const v8h*)(qp + 48);
#pragma unroll
      for (int e = 0; e < 8; ++e) {
        a0[e] = l0[e]; a0[e + 8] = h0[e];
        a1[e] = l1[e]; a1[e + 8] = h1[e];
      }
    }
    v8f o[4];
#pragma unroll
    for (int j = 0; j < 4; ++j) o[j] = v8f{0.f,0.f,0.f,0.f,0.f,0.f,0.f,0.f};
    float rsum = 0.f;
    int kcMax = (q0 + 15) >> 5;
    for (int kc = 0; kc <= kcMax; ++kc) {
#pragma unroll
      for (int t2 = 0; t2 < 2; ++t2) {       // two 16-key score sub-tiles
        int n0 = kc * 32 + t2 * 16;
        const _Float16* kp = kh + (n0 + row) * DHMM + 8 * hi;
        v8h b0a = *(const v8h*)(kp);
        v8h b0b = *(const v8h*)(kp + 16);
        v8h b1a = *(const v8h*)(kp + 32);
        v8h b1b = *(const v8h*)(kp + 48);
        v16h b0, b1;
#pragma unroll
        for (int e = 0; e < 8; ++e) {
          b0[e] = b0a[e]; b0[e + 8] = b0b[e];
          b1[e] = b1a[e]; b1[e + 8] = b1b[e];
        }
        v8f sc = {0.f,0.f,0.f,0.f,0.f,0.f,0.f,0.f};
        sc = __builtin_amdgcn_wmma_f32_16x16x32_f16(false, a0, false, b0,
                                                    (short)0, sc, false, false);
        sc = __builtin_amdgcn_wmma_f32_16x16x32_f16(false, a1, false, b1,
                                                    (short)0, sc, false, false);
        int kk = n0 + row;
        float uk = uls[kk];
#pragma unroll
        for (int vv = 0; vv < 8; ++vv) {
          int qq = q0 + vv + hi * 8;
          float c = (kk <= qq) ? sc[vv] * 0.125f * expf(uk - pls[qq]) : 0.f;
          mystg[(vv + hi * 8) * 32 + t2 * 16 + row] = (_Float16)c;
        }
      }
      asm volatile("s_wait_dscnt 0" ::: "memory");   // stage visible to wave
      v16h ac;
      float part = 0.f;
      {
        const _Float16* sp = mystg + row * 32 + 8 * hi;
        v8h cl = *(const v8h*)(sp);
        v8h ch = *(const v8h*)(sp + 16);
#pragma unroll
        for (int e = 0; e < 8; ++e) {
          ac[e] = cl[e]; ac[e + 8] = ch[e];
          part += (float)cl[e] + (float)ch[e];
        }
      }
      rsum += part;
#pragma unroll
      for (int j = 0; j < 4; ++j) {
        const _Float16* vp = vh + (j * 16 + row) * SS + kc * 32 + 8 * hi;
        v8h va = *(const v8h*)(vp);
        v8h vb2 = *(const v8h*)(vp + 16);
        v16h bv;
#pragma unroll
        for (int e = 0; e < 8; ++e) { bv[e] = va[e]; bv[e + 8] = vb2[e]; }
        o[j] = __builtin_amdgcn_wmma_f32_16x16x32_f16(false, ac, false, bv,
                                                      (short)0, o[j], false, false);
      }
    }
    rsum += __shfl_xor(rsum, 16, 32);        // full row sum (row = lane&15)
    float nval = fmaxf(fabsf(rsum), els[q0 + row]) + 1e-6f;
#pragma unroll
    for (int j = 0; j < 4; ++j) {
#pragma unroll
      for (int vv = 0; vv < 8; ++vv) {
        float nn = __shfl(nval, vv + 8 * hi, 32);
        int qq = q0 + vv + hi * 8;
        hm[((size_t)(b * SS + qq)) * DIE + h * DHMM + j * 16 + row] = o[j][vv] / nn;
      }
    }
  }
}

// ---------------------------------------------------------------------------
__global__ void k_mpost(const float* __restrict__ hm, const float* __restrict__ onw,
                        const float* __restrict__ skip, const float* __restrict__ xca,
                        const float* __restrict__ up, float* __restrict__ out) {
  __shared__ float s1[DIE], s2[DIE];
  int row = blockIdx.x, t = threadIdx.x;
  float vv = hm[(size_t)row * DIE + t];
  s1[t] = vv; s2[t] = vv * vv;
  __syncthreads();
  for (int off = 32; off > 0; off >>= 1) {
    if ((t & 63) < off) { s1[t] += s1[t + off]; s2[t] += s2[t + off]; }
    __syncthreads();
  }
  int hb = t & ~63;
  float mu = s1[hb] * (1.0f / 64.0f);
  float var = s2[hb] * (1.0f / 64.0f) - mu * mu;
  float xn = (vv - mu) * rsqrtf(var + 1e-5f) * onw[t];
  float z = up[(size_t)row * 512 + 256 + t];
  out[(size_t)row * DIE + t] =
      (xn + skip[t] * xca[(size_t)row * DIE + t]) * dsilu(z);
}

// ---------------------------------------------------------------------------
__global__ void k_ghn(const float* __restrict__ y, const float* __restrict__ w,
                      float* __restrict__ h) {
  __shared__ float s1[DD], s2[DD];
  int row = blockIdx.x, t = threadIdx.x;
  float vv = y[(size_t)row * DD + t];
  s1[t] = vv; s2[t] = vv * vv;
  __syncthreads();
  for (int off = 16; off > 0; off >>= 1) {
    if ((t & 31) < off) { s1[t] += s1[t + off]; s2[t] += s2[t + off]; }
    __syncthreads();
  }
  int hb = t & ~31;
  float mu = s1[hb] * (1.0f / 32.0f);
  float var = s2[hb] * (1.0f / 32.0f) - mu * mu;
  h[(size_t)row * DD + t] += (vv - mu) * rsqrtf(var + 1e-5f) * w[t];
}

// ---------------------------------------------------------------------------
__global__ void __launch_bounds__(128)
k_sscan(const float* __restrict__ gi, const float* __restrict__ gf,
        const float* __restrict__ gz, const float* __restrict__ go,
        const float* __restrict__ recw, const float* __restrict__ bias,
        float* __restrict__ y) {
  __shared__ float hls[DHSS];
  __shared__ float rls[4 * DHSS];
  int bh = blockIdx.x, b = bh >> 2, h = bh & 3;
  int t = threadIdx.x;
  const float* rw = recw + (size_t)h * DHSS * 4 * DHSS;   // (32,128)
  const float* bb = bias + (size_t)h * 4 * DHSS;
  float cst = 0.f, nst = 0.f, mst = 0.f;
  if (t < DHSS) hls[t] = 0.f;
  __syncthreads();
  for (int s = 0; s < SS; ++s) {
    float acc = 0.f;
#pragma unroll
    for (int d = 0; d < DHSS; ++d) acc += hls[d] * rw[d * 4 * DHSS + t];
    rls[t] = acc;
    __syncthreads();
    if (t < DHSS) {
      size_t g = ((size_t)(b * SS + s)) * DD + h * DHSS + t;
      float iraw = gi[g] + rls[t]             + bb[t];
      float fraw = gf[g] + rls[DHSS + t]      + bb[DHSS + t];
      float zraw = gz[g] + rls[2 * DHSS + t]  + bb[2 * DHSS + t];
      float oraw = go[g] + rls[3 * DHSS + t]  + bb[3 * DHSS + t];
      float lfm = mst + dlogsig(fraw);
      float mn = fmaxf(iraw, lfm);
      float ig = expf(iraw - mn);
      float fg = expf(lfm - mn);
      cst = fg * cst + ig * tanhf(zraw);
      nst = fg * nst + ig;
      mst = mn;
      float hn = dsigm(oraw) * cst / nst;
      y[g] = hn;
      hls[t] = hn;
    }
    __syncthreads();
  }
}

// ---------------------------------------------------------------------------
__global__ void k_glu(const float* __restrict__ up, float* __restrict__ out) {
  int i = blockIdx.x * blockDim.x + threadIdx.x;
  if (i >= ROWS * DIE) return;
  int row = i / DIE, c = i - row * DIE;
  float g = up[(size_t)row * 512 + c];
  float u2 = up[(size_t)row * 512 + 256 + c];
  out[i] = dgelu(g) * u2;
}

// ---------------------------------------------------------------------------
extern "C" void kernel_launch(void* const* d_in, const int* in_sizes, int n_in,
                              void* d_out, int out_size, void* d_ws, size_t ws_size,
                              hipStream_t stream) {
  const int*   x        = (const int*)  d_in[0];
  const float* emb      = (const float*)d_in[1];
  const float* m_ln_w   = (const float*)d_in[2];
  const float* m_up_w   = (const float*)d_in[3];
  const float* m_conv_w = (const float*)d_in[4];
  const float* m_conv_b = (const float*)d_in[5];
  const float* m_q_w    = (const float*)d_in[6];
  const float* m_k_w    = (const float*)d_in[7];
  const float* m_v_w    = (const float*)d_in[8];
  const float* m_ig_w   = (const float*)d_in[9];
  const float* m_ig_b   = (const float*)d_in[10];
  const float* m_fg_w   = (const float*)d_in[11];
  const float* m_fg_b   = (const float*)d_in[12];
  const float* m_on_w   = (const float*)d_in[13];
  const float* m_skip   = (const float*)d_in[14];
  const float* m_down_w = (const float*)d_in[15];
  const float* s_ln1_w  = (const float*)d_in[16];
  const float* s_conv_w = (const float*)d_in[17];
  const float* s_conv_b = (const float*)d_in[18];
  const float* s_ig_w   = (const float*)d_in[19];
  const float* s_fg_w   = (const float*)d_in[20];
  const float* s_zg_w   = (const float*)d_in[21];
  const float* s_og_w   = (const float*)d_in[22];
  const float* s_rec_w  = (const float*)d_in[23];
  const float* s_bias   = (const float*)d_in[24];
  const float* s_gn_w   = (const float*)d_in[25];
  const float* s_ln2_w  = (const float*)d_in[26];
  const float* s_ff_up  = (const float*)d_in[27];
  const float* s_ff_dn  = (const float*)d_in[28];
  const float* post_w   = (const float*)d_in[29];

  float* W = (float*)d_ws;
  size_t o = 0;
  float* h   = W + o; o += (size_t)ROWS * DD;    // residual stream
  float* xn  = W + o; o += (size_t)ROWS * DD;    // LN output (reused)
  float* up  = W + o; o += (size_t)ROWS * 512;   // up-proj / FFN-up output
  float* xca = W + o; o += (size_t)ROWS * DIE;   // conv output (reused as sLSTM xc)
  float* qb  = W + o; o += (size_t)ROWS * DIE;   // q   (reused: mpost out / gi / glu out)
  float* kb  = W + o; o += (size_t)ROWS * DIE;   // k   (reused: gf+gz)
  float* vb  = W + o; o += (size_t)ROWS * DIE;   // v   (reused: go)
  float* hm  = W + o; o += (size_t)ROWS * DIE;   // attn out (reused as y)
  float* ip  = W + o; o += (size_t)BB * NHH * SS;
  float* fp  = W + o; o += (size_t)BB * NHH * SS;
  float* uu  = W + o; o += (size_t)BB * NHH * SS;
  float* pmv = W + o; o += (size_t)BB * NHH * SS;
  float* emv = W + o; o += (size_t)BB * NHH * SS;
  _Float16* q16 = (_Float16*)(W + o); o += (size_t)ROWS * DIE / 2;
  _Float16* k16 = (_Float16*)(W + o); o += (size_t)ROWS * DIE / 2;
  _Float16* vt  = (_Float16*)(W + o); o += (size_t)ROWS * DIE / 2;
  (void)ws_size; (void)n_in; (void)in_sizes; (void)out_size;

  const int T = 256;
  // ---- mLSTM block ----
  k_embed<<<(ROWS*DD + T-1)/T, T, 0, stream>>>(x, emb, h);
  k_ln<<<ROWS, DD, 0, stream>>>(h, m_ln_w, xn);
  k_gemm<0><<<(ROWS/16)*(512/16)/8, T, 0, stream>>>(xn, m_up_w, up, ROWS, 512, DD);
  k_conv_silu<<<(ROWS*DIE + T-1)/T, T, 0, stream>>>(up, 512, m_conv_w, m_conv_b, xca, DIE);
  k_blockdiag<<<(ROWS*DIE + T-1)/T, T, 0, stream>>>(xca, DIE, m_q_w, 8, qb, DIE);
  k_blockdiag<<<(ROWS*DIE + T-1)/T, T, 0, stream>>>(xca, DIE, m_k_w, 8, kb, DIE);
  k_blockdiag<<<(ROWS*DIE + T-1)/T, T, 0, stream>>>(up, 512, m_v_w, 8, vb, DIE);
  k_mgate<<<(ROWS*NHH*2 + T-1)/T, T, 0, stream>>>(qb, kb, vb, m_ig_w, m_ig_b,
                                                  m_fg_w, m_fg_b, ip, fp);
  k_tof16<<<(ROWS*DIE + T-1)/T, T, 0, stream>>>(qb, kb, vb, q16, k16, vt);
  k_mscan<<<(BB*NHH + 63)/64, 64, 0, stream>>>(ip, fp, uu, pmv, emv);
  size_t attnShm = (size_t)3*SS*DHMM*2 + 3*SS*4 + 8*16*32*2;   // 109568 B
  k_attn<<<BB*NHH, 256, attnShm, stream>>>(q16, k16, vt, uu, pmv, emv, hm);
  k_mpost<<<ROWS, DIE, 0, stream>>>(hm, m_on_w, m_skip, xca, up, qb);
  k_gemm<1><<<(ROWS/16)*(DD/16)/8, T, 0, stream>>>(qb, m_down_w, h, ROWS, DD, DIE);

  // ---- sLSTM blocks ----
  float* gi = qb;
  float* gf = kb;
  float* gz = kb + (size_t)ROWS * DD;
  float* go = vb;
  float* xc = xca;
  float* y  = hm;
  for (int i = 0; i < 2; ++i) {
    k_ln<<<ROWS, DD, 0, stream>>>(h, s_ln1_w + (size_t)i*DD, xn);
    k_conv_silu<<<(ROWS*DD + T-1)/T, T, 0, stream>>>(xn, DD, s_conv_w + (size_t)i*DD*KCONV,
                                                     s_conv_b + (size_t)i*DD, xc, DD);
    size_t wOff = (size_t)i * NHH * DHSS * DHSS;
    k_blockdiag<<<(ROWS*DD + T-1)/T, T, 0, stream>>>(xc, DD, s_ig_w + wOff, DHSS, gi, DD);
    k_blockdiag<<<(ROWS*DD + T-1)/T, T, 0, stream>>>(xc, DD, s_fg_w + wOff, DHSS, gf, DD);
    k_blockdiag<<<(ROWS*DD + T-1)/T, T, 0, stream>>>(xn, DD, s_zg_w + wOff, DHSS, gz, DD);
    k_blockdiag<<<(ROWS*DD + T-1)/T, T, 0, stream>>>(xn, DD, s_og_w + wOff, DHSS, go, DD);
    k_sscan<<<BB*NHH, 128, 0, stream>>>(gi, gf, gz, go,
                                        s_rec_w + (size_t)i*NHH*DHSS*4*DHSS,
                                        s_bias  + (size_t)i*NHH*4*DHSS, y);
    k_ghn<<<ROWS, DD, 0, stream>>>(y, s_gn_w + (size_t)i*DD, h);
    k_ln<<<ROWS, DD, 0, stream>>>(h, s_ln2_w + (size_t)i*DD, xn);
    k_gemm<0><<<(ROWS/16)*(512/16)/8, T, 0, stream>>>(xn, s_ff_up + (size_t)i*512*DD,
                                                      up, ROWS, 512, DD);
    k_glu<<<(ROWS*DIE + T-1)/T, T, 0, stream>>>(up, qb);
    k_gemm<1><<<(ROWS/16)*(DD/16)/8, T, 0, stream>>>(qb, s_ff_dn + (size_t)i*DD*DIE,
                                                     h, ROWS, DD, DIE);
  }
  k_ln<<<ROWS, DD, 0, stream>>>(h, post_w, (float*)d_out);
}
